// FlowN_66219805770046
// MI455X (gfx1250) — compile-verified, matrix operand order checked by reading, not run on hardware
//
#include <hip/hip_runtime.h>
#include <math.h>

// ---------------- problem constants ----------------
#define BB   2
#define CC   96
#define HH   48
#define WWD  48
#define LL   (HH*WWD)        // 2304
#define DD   192             // inner VSS dim (2*C)
#define KK   4               // scan directions
#define RNK  6
#define NST  16              // state size
#define XPN  (RNK + 2*NST)   // 38
#define RP   48              // padded x_dbl row stride
#define NHh  8
#define DHh  12
#define ROWS (BB*LL)         // 4608
#define NTOT (BB*CC*LL)      // 442368
#define FLOWN (BB*2*LL)      // 9216

// ---------------- WMMA types ----------------
typedef _Float16 v8h  __attribute__((ext_vector_type(8)));
typedef _Float16 v16h __attribute__((ext_vector_type(16)));
typedef float    v8f  __attribute__((ext_vector_type(8)));

__device__ __forceinline__ float geluf(float x) {
    float x3 = x * x * x;
    return 0.5f * x * (1.0f + tanhf(0.7978845608028654f * (x + 0.044715f * x3)));
}
__device__ __forceinline__ float siluf(float x) { return x / (1.0f + expf(-x)); }

// =====================================================================
// WMMA GEMM:  Y[M,N] = act( X[M,K] @ Wt[N,K]^T + bias + resid )
// One wave = 16x16 f32 tile (v_wmma_f32_16x16x32_f16), 8 waves/block =
// 64x32 block tile. Double-buffered LDS staging: next K-slice's global
// loads are issued before the WMMA consumes the current slice; one
// barrier per K-step. Fast path = vectorized loads + packed cvt + one
// aligned b128 LDS store; edge path = clamped branchless loads (cndmask,
// no exec-mask branching, no per-element waits).
// =====================================================================
#define TM 64
#define TN 32
#define TK 32
#define LPITCH 40   // halves per LDS row (80B: 16B aligned, conflict-free)

__global__ __launch_bounds__(256)
void k_gemm(const float* __restrict__ X, int ldx,
            const float* __restrict__ Wt, int ldw,
            const float* __restrict__ bias,
            const float* __restrict__ resid, int ldr,
            float* __restrict__ Y, int ldy,
            int M, int N, int Kd, int act)
{
    __shared__ __align__(16) _Float16 As[2][TM][LPITCH];
    __shared__ __align__(16) _Float16 Bs[2][TN][LPITCH];

    const int tid  = threadIdx.x;
    const int wave = tid >> 5;
    const int lane = tid & 31;
    const int m0 = blockIdx.y * TM;
    const int n0 = blockIdx.x * TN;
    const int mi = wave >> 1;          // 0..3 (M sub-tile)
    const int ni = wave & 1;           // 0..1 (N sub-tile)
    const int mrow  = lane & 15;
    const int khalf = (lane >> 4) * 8; // ISA 16-bit A/B layout K split

    v8f acc = {0.f,0.f,0.f,0.f,0.f,0.f,0.f,0.f};

    const int ar = tid >> 2;             // 0..63  (A row)
    const int c0 = (tid & 3) * 8;        // 0,8,16,24
    const int gm = m0 + ar;
    const int gmc = (gm < M) ? gm : (M - 1);
    const int br  = (tid & 127) >> 2;    // 0..31  (B row, tid<128)
    const int gn  = n0 + br;
    const int gnc = (gn < N) ? gn : (N - 1);

    auto stageA = [&](int bufi, int k0) {
        v8h hv;
        if (gm < M && k0 + c0 + 8 <= Kd) {
            const float* xp = X + (size_t)gm * ldx + k0 + c0;
            __builtin_prefetch(xp + TK, 0, 3);
            #pragma unroll
            for (int i = 0; i < 8; ++i) hv[i] = (_Float16)xp[i];
        } else {
            const float* xr = X + (size_t)gmc * ldx;
            #pragma unroll
            for (int i = 0; i < 8; ++i) {
                int kk = k0 + c0 + i;
                int kc = (kk < Kd) ? kk : (Kd - 1);
                float v = xr[kc];                     // clamped, always in-bounds
                hv[i] = (_Float16)((gm < M && kk < Kd) ? v : 0.f);
            }
        }
        *(v8h*)&As[bufi][ar][c0] = hv;
    };
    auto stageB = [&](int bufi, int k0) {
        if (tid >= 128) return;
        v8h hv;
        if (gn < N && k0 + c0 + 8 <= Kd) {
            const float* wp = Wt + (size_t)gn * ldw + k0 + c0;
            __builtin_prefetch(wp + TK, 0, 3);
            #pragma unroll
            for (int i = 0; i < 8; ++i) hv[i] = (_Float16)wp[i];
        } else {
            const float* wr = Wt + (size_t)gnc * ldw;
            #pragma unroll
            for (int i = 0; i < 8; ++i) {
                int kk = k0 + c0 + i;
                int kc = (kk < Kd) ? kk : (Kd - 1);
                float v = wr[kc];
                hv[i] = (_Float16)((gn < N && kk < Kd) ? v : 0.f);
            }
        }
        *(v8h*)&Bs[bufi][br][c0] = hv;
    };

    stageA(0, 0);
    stageB(0, 0);
    int buf = 0;
    for (int k0 = 0; k0 < Kd; k0 += TK, buf ^= 1) {
        __syncthreads();                       // staging of `buf` complete
        if (k0 + TK < Kd) {                    // issue next slice early
            stageA(buf ^ 1, k0 + TK);
            stageB(buf ^ 1, k0 + TK);
        }
        // ---- assemble fragments per ISA 16-bit layout ----
        const _Float16* ap = &As[buf][mi * 16 + mrow][khalf];
        const _Float16* bp = &Bs[buf][ni * 16 + mrow][khalf];
        v8h alo = *(const v8h*)ap;
        v8h ahi = *(const v8h*)(ap + 16);
        v8h blo = *(const v8h*)bp;
        v8h bhi = *(const v8h*)(bp + 16);
        v16h a = __builtin_shufflevector(alo, ahi, 0,1,2,3,4,5,6,7,8,9,10,11,12,13,14,15);
        v16h b = __builtin_shufflevector(blo, bhi, 0,1,2,3,4,5,6,7,8,9,10,11,12,13,14,15);

        acc = __builtin_amdgcn_wmma_f32_16x16x32_f16(
                  false, a, false, b, (short)0, acc, false, false);
    }

    // ---- write D: VGPR r -> M = r + 8*(lane>=16), N = lane&15 ----
    const int col = n0 + ni * 16 + (lane & 15);
    const int rb  = m0 + mi * 16 + (lane >> 4) * 8;
    float bv = (bias && col < N) ? bias[col] : 0.0f;
    #pragma unroll
    for (int r = 0; r < 8; ++r) {
        int row = rb + r;
        if (row < M && col < N) {
            float v = acc[r] + bv;
            if (resid) v += resid[(size_t)row * ldr + col];
            if (act == 1) v = geluf(v);
            Y[(size_t)row * ldy + col] = v;
        }
    }
}

// =====================================================================
// LayerNorm over last dim, one wave per row (wave32 butterfly reduce)
// =====================================================================
__global__ void k_ln(const float* __restrict__ x, const float* __restrict__ w,
                     const float* __restrict__ b, float* __restrict__ y,
                     int rows, int Cd)
{
    int wid  = (blockIdx.x * blockDim.x + threadIdx.x) >> 5;
    int lane = threadIdx.x & 31;
    if (wid >= rows) return;
    const float* xr = x + (size_t)wid * Cd;
    float s = 0.f;
    for (int c = lane; c < Cd; c += 32) s += xr[c];
    #pragma unroll
    for (int off = 16; off > 0; off >>= 1) s += __shfl_xor(s, off, 32);
    float mu = s / (float)Cd;
    float v = 0.f;
    for (int c = lane; c < Cd; c += 32) { float d = xr[c] - mu; v += d * d; }
    #pragma unroll
    for (int off = 16; off > 0; off >>= 1) v += __shfl_xor(v, off, 32);
    float rstd = rsqrtf(v / (float)Cd + 1e-5f);
    float* yr = y + (size_t)wid * Cd;
    for (int c = lane; c < Cd; c += 32)
        yr[c] = (xr[c] - mu) * rstd * w[c] + b[c];
}

// =====================================================================
// Depthwise 3x3 conv, NCHW, SAME, optional SiLU
// =====================================================================
__global__ void k_dwconv(const float* __restrict__ x, const float* __restrict__ w,
                         const float* __restrict__ b, float* __restrict__ y,
                         int Bn, int Cd, int dosilu)
{
    int idx = blockIdx.x * blockDim.x + threadIdx.x;
    int tot = Bn * Cd * LL;
    if (idx >= tot) return;
    int wq = idx % WWD; int t = idx / WWD;
    int hq = t % HH;    t /= HH;
    int c  = t % Cd;    int bb = t / Cd;
    const float* xc = x + ((size_t)bb * Cd + c) * LL;
    const float* wc = w + c * 9;
    float s = b[c];
    #pragma unroll
    for (int dy = -1; dy <= 1; ++dy)
        #pragma unroll
        for (int dx = -1; dx <= 1; ++dx) {
            int hy = hq + dy, wx = wq + dx;
            if (hy >= 0 && hy < HH && wx >= 0 && wx < WWD)
                s += wc[(dy + 1) * 3 + (dx + 1)] * xc[hy * WWD + wx];
        }
    if (dosilu) s = siluf(s);
    y[idx] = s;
}

// ---------------- layout shuffles ----------------
// NCHW -> NHWC
__global__ void k_n2h(const float* __restrict__ src, float* __restrict__ dst,
                      int Bn, int Cd)
{
    int idx = blockIdx.x * blockDim.x + threadIdx.x;
    int tot = Bn * Cd * LL;
    if (idx >= tot) return;
    int l = idx % LL; int t = idx / LL; int c = t % Cd; int b = t / Cd;
    dst[((size_t)b * LL + l) * Cd + c] = src[idx];
}
// NHWC(strided,offset) -> NCHW, optional gelu / accumulate
__global__ void k_h2n(const float* __restrict__ src, float* __restrict__ dst,
                      int Bn, int Cd, int lds, int off, int act, int add)
{
    int idx = blockIdx.x * blockDim.x + threadIdx.x;
    int tot = Bn * Cd * LL;
    if (idx >= tot) return;
    int l = idx % LL; int t = idx / LL; int c = t % Cd; int b = t / Cd;
    float v = src[((size_t)b * LL + l) * lds + off + c];
    if (act == 1) v = geluf(v);
    size_t di = ((size_t)b * Cd + c) * LL + l;
    if (add) dst[di] += v; else dst[di] = v;
}

// build 4-direction xs, l-major: xst (B,4,L,D) from xp2 (B,D,L) NCHW
__global__ void k_build_xst(const float* __restrict__ xp, float* __restrict__ xst)
{
    int idx = blockIdx.x * blockDim.x + threadIdx.x;
    int tot = BB * LL * DD;
    if (idx >= tot) return;
    int d = idx % DD; int t = idx / DD; int l = t % LL; int b = t / LL;
    int h = l / WWD, w = l % WWD;
    int lt = w * HH + h;                       // transposed scan order
    float v = xp[((size_t)b * DD + d) * LL + l];
    xst[(((size_t)b * KK + 0) * LL + l)            * DD + d] = v;
    xst[(((size_t)b * KK + 1) * LL + lt)           * DD + d] = v;
    xst[(((size_t)b * KK + 2) * LL + (LL - 1 - l)) * DD + d] = v;
    xst[(((size_t)b * KK + 3) * LL + (LL - 1 - lt))* DD + d] = v;
}

// =====================================================================
// Selective scan: lane per (b,k,d); 16-state in VGPRs; l-major coalesced
// =====================================================================
__global__ void k_scan(const float* __restrict__ u,   // xst (B,K,L,D)
                       const float* __restrict__ dtd, // (B,K,L,D)
                       const float* __restrict__ dt_bias,
                       const float* __restrict__ A_logs,
                       const float* __restrict__ xdb, // (B,K,L,RP) cols 6..38 = B|C
                       const float* __restrict__ Ds,
                       float* __restrict__ ys)        // (B,K,L,D)
{
    int idx = blockIdx.x * blockDim.x + threadIdx.x;
    if (idx >= BB * KK * DD) return;
    int d = idx % DD; int t = idx / DD; int k = t % KK; int b = t / KK;
    int kd = k * DD + d;
    float A[NST];
    #pragma unroll
    for (int n = 0; n < NST; ++n) A[n] = -expf(A_logs[(size_t)kd * NST + n]);
    float dtb = dt_bias[kd];
    float Dk  = Ds[kd];
    float h[NST];
    #pragma unroll
    for (int n = 0; n < NST; ++n) h[n] = 0.f;
    size_t rowbase = ((size_t)b * KK + k) * LL;
    for (int l = 0; l < LL; ++l) {
        size_t r = rowbase + l;
        float ut = u[r * DD + d];
        float dv = dtd[r * DD + d] + dtb;
        float dt = (dv > 20.f) ? dv : log1pf(expf(dv));   // softplus
        const float* xr = xdb + r * RP;
        float du = dt * ut;
        float y = 0.f;
        #pragma unroll
        for (int n = 0; n < NST; ++n) {
            float dA = expf(dt * A[n]);
            h[n] = dA * h[n] + du * xr[RNK + n];
            y += h[n] * xr[RNK + NST + n];
        }
        ys[r * DD + d] = y + Dk * ut;
    }
}

// combine 4 scan directions -> yc (B,L,D)
__global__ void k_combine(const float* __restrict__ ys, float* __restrict__ yc)
{
    int idx = blockIdx.x * blockDim.x + threadIdx.x;
    int tot = BB * LL * DD;
    if (idx >= tot) return;
    int d = idx % DD; int t = idx / DD; int l = t % LL; int b = t / LL;
    int h = l / WWD, w = l % WWD;
    int lt = w * HH + h;
    size_t b0 = (size_t)b * KK * LL;
    float v = ys[(b0 + 0 * LL + l)              * DD + d]
            + ys[(b0 + 2 * LL + (LL - 1 - l))   * DD + d]
            + ys[(b0 + 1 * LL + lt)             * DD + d]
            + ys[(b0 + 3 * LL + (LL - 1 - lt))  * DD + d];
    yc[((size_t)b * LL + l) * DD + d] = v;
}

// y *= silu(z), z strided (the in_proj gate half)
__global__ void k_gate(float* __restrict__ y, const float* __restrict__ z,
                       int rows, int Cd, int ldz, int zoff)
{
    int idx = blockIdx.x * blockDim.x + threadIdx.x;
    int tot = rows * Cd;
    if (idx >= tot) return;
    int c = idx % Cd; int r = idx / Cd;
    y[idx] *= siluf(z[(size_t)r * ldz + zoff + c]);
}

__global__ void k_add(const float* __restrict__ a, const float* __restrict__ b,
                      float* __restrict__ c, int n)
{
    int i = blockIdx.x * blockDim.x + threadIdx.x;
    if (i < n) c[i] = a[i] + b[i];
}
__global__ void k_copy(const float* __restrict__ a, float* __restrict__ b, int n)
{
    int i = blockIdx.x * blockDim.x + threadIdx.x;
    if (i < n) b[i] = a[i];
}

// tiny cross-"attention": seq axis = batch dim (S=2), per (l,h) 2x2 softmax
__global__ void k_mha(const float* __restrict__ q, const float* __restrict__ k,
                      const float* __restrict__ v, float* __restrict__ o)
{
    int idx = blockIdx.x * blockDim.x + threadIdx.x;
    if (idx >= LL * NHh) return;
    int h = idx % NHh; int l = idx / NHh;
    const float scale = rsqrtf((float)DHh);
    float sc[2][2];
    #pragma unroll
    for (int s = 0; s < 2; ++s)
        #pragma unroll
        for (int tt = 0; tt < 2; ++tt) {
            const float* qp = q + ((size_t)s * LL + l) * CC + h * DHh;
            const float* kp = k + ((size_t)tt * LL + l) * CC + h * DHh;
            float dsum = 0.f;
            #pragma unroll
            for (int j = 0; j < DHh; ++j) dsum += qp[j] * kp[j];
            sc[s][tt] = dsum * scale;
        }
    #pragma unroll
    for (int s = 0; s < 2; ++s) {
        float m = fmaxf(sc[s][0], sc[s][1]);
        float e0 = expf(sc[s][0] - m), e1 = expf(sc[s][1] - m);
        float inv = 1.f / (e0 + e1);
        float a0 = e0 * inv, a1 = e1 * inv;
        const float* v0 = v + ((size_t)0 * LL + l) * CC + h * DHh;
        const float* v1 = v + ((size_t)1 * LL + l) * CC + h * DHh;
        float* op = o + ((size_t)s * LL + l) * CC + h * DHh;
        #pragma unroll
        for (int j = 0; j < DHh; ++j) op[j] = a0 * v0[j] + a1 * v1[j];
    }
}

// flow conv: 3x3 over concat([x2a, f1]) -> (B,2,H,W), no bias
__global__ void k_flowconv(const float* __restrict__ in0, const float* __restrict__ in1,
                           const float* __restrict__ wgt, float* __restrict__ out)
{
    int idx = blockIdx.x * blockDim.x + threadIdx.x;
    if (idx >= BB * 2 * LL) return;
    int w = idx % WWD; int t = idx / WWD;
    int h = t % HH;    t /= HH;
    int oc = t % 2;    int b = t / 2;
    float s = 0.f;
    for (int ic = 0; ic < 2 * CC; ++ic) {
        const float* src = (ic < CC) ? in0 + ((size_t)b * CC + ic) * LL
                                     : in1 + ((size_t)b * CC + (ic - CC)) * LL;
        const float* wp = wgt + ((size_t)oc * 2 * CC + ic) * 9;
        #pragma unroll
        for (int dy = -1; dy <= 1; ++dy)
            #pragma unroll
            for (int dx = -1; dx <= 1; ++dx) {
                int hy = h + dy, wx = w + dx;
                if (hy >= 0 && hy < HH && wx >= 0 && wx < WWD)
                    s += wp[(dy + 1) * 3 + (dx + 1)] * src[hy * WWD + wx];
            }
    }
    out[idx] = s;
}

// bilinear grid sample (zeros pad, align_corners=False) with flow offset
__global__ void k_warp(const float* __restrict__ img, const float* __restrict__ flow,
                       float* __restrict__ out)
{
    int idx = blockIdx.x * blockDim.x + threadIdx.x;
    if (idx >= BB * LL) return;
    int w = idx % WWD; int t = idx / WWD;
    int h = t % HH;    int b = t / HH;
    float fx = flow[((size_t)b * 2 + 0) * LL + h * WWD + w];
    float fy = flow[((size_t)b * 2 + 1) * LL + h * WWD + w];
    float gxn = -1.f + 2.f * (float)w / (float)(WWD - 1) + fx / (float)WWD;
    float gyn = -1.f + 2.f * (float)h / (float)(HH - 1)  + fy / (float)HH;
    float gx = (gxn + 1.f) * (float)WWD * 0.5f - 0.5f;
    float gy = (gyn + 1.f) * (float)HH  * 0.5f - 0.5f;
    float x0f = floorf(gx), y0f = floorf(gy);
    float wx1 = gx - x0f, wx0 = 1.f - wx1;
    float wy1 = gy - y0f, wy0 = 1.f - wy1;
    int xi[4], yi[4]; float wt[4];
    float xs[2] = {x0f, x0f + 1.f}, yss[2] = {y0f, y0f + 1.f};
    float wxs[2] = {wx0, wx1},      wys[2] = {wy0, wy1};
    #pragma unroll
    for (int cy = 0; cy < 2; ++cy)
        #pragma unroll
        for (int cx = 0; cx < 2; ++cx) {
            int q = cy * 2 + cx;
            float X = xs[cx], Y = yss[cy];
            bool valid = (X >= 0.f) && (X <= (float)(WWD - 1)) &&
                         (Y >= 0.f) && (Y <= (float)(HH - 1));
            xi[q] = min(max((int)X, 0), WWD - 1);
            yi[q] = min(max((int)Y, 0), HH - 1);
            wt[q] = valid ? (wxs[cx] * wys[cy]) : 0.f;
        }
    for (int c = 0; c < CC; ++c) {
        const float* ip = img + ((size_t)b * CC + c) * LL;
        float acc = 0.f;
        #pragma unroll
        for (int q = 0; q < 4; ++q) acc += wt[q] * ip[yi[q] * WWD + xi[q]];
        out[((size_t)b * CC + c) * LL + h * WWD + w] = acc;
    }
}

// =====================================================================
// Host-side orchestration
// =====================================================================
static inline void gemm(hipStream_t s, const float* X, int ldx,
                        const float* Wt, int ldw, const float* bias,
                        const float* resid, int ldr,
                        float* Y, int ldy, int M, int N, int Kd, int act)
{
    dim3 g((unsigned)((N + TN - 1) / TN), (unsigned)((M + TM - 1) / TM));
    k_gemm<<<g, 256, 0, s>>>(X, ldx, Wt, ldw, bias, resid, ldr, Y, ldy, M, N, Kd, act);
}
#define GRID1(n) dim3((unsigned)(((n) + 255) / 256)), dim3(256)

struct VssW {
    const float *ln_w, *ln_b, *in_proj_w, *conv_w, *conv_b, *x_proj_w,
                *dt_proj_w, *dt_bias, *A_logs, *Ds, *on_w, *on_b, *out_proj_w;
};
static VssW vss_at(void* const* d_in, int base) {
    VssW w;
    w.ln_w       = (const float*)d_in[base + 0];
    w.ln_b       = (const float*)d_in[base + 1];
    w.in_proj_w  = (const float*)d_in[base + 2];
    w.conv_w     = (const float*)d_in[base + 3];
    w.conv_b     = (const float*)d_in[base + 4];
    w.x_proj_w   = (const float*)d_in[base + 5];
    w.dt_proj_w  = (const float*)d_in[base + 6];
    w.dt_bias    = (const float*)d_in[base + 7];
    w.A_logs     = (const float*)d_in[base + 8];
    w.Ds         = (const float*)d_in[base + 9];
    w.on_w       = (const float*)d_in[base + 10];
    w.on_b       = (const float*)d_in[base + 11];
    w.out_proj_w = (const float*)d_in[base + 12];
    return w;
}

// VSS scratch layout sizes (floats)
#define SZ_XN  ((size_t)ROWS * CC)
#define SZ_XZ  ((size_t)ROWS * 2 * DD)
#define SZ_XP  ((size_t)BB * DD * LL)
#define SZ_XST ((size_t)BB * KK * LL * DD)
#define SZ_XDB ((size_t)BB * KK * LL * RP)
#define SZ_YC  ((size_t)BB * LL * DD)

static void run_vss(hipStream_t st, const float* in, float* out, const VssW& w, float* S, int final_act)
{
    float* xn  = S;
    float* xz  = xn  + SZ_XN;
    float* xp  = xz  + SZ_XZ;
    float* xp2 = xp  + SZ_XP;
    float* xst = xp2 + SZ_XP;
    float* xdb = xst + SZ_XST;
    float* dtd = xdb + SZ_XDB;
    float* ys  = dtd + SZ_XST;
    float* yc  = ys  + SZ_XST;
    float* yl  = yc  + SZ_YC;

    k_ln<<<dim3((ROWS + 7) / 8), dim3(256), 0, st>>>(in, w.ln_w, w.ln_b, xn, ROWS, CC);
    gemm(st, xn, CC, w.in_proj_w, CC, nullptr, nullptr, 0, xz, 2 * DD, ROWS, 2 * DD, CC, 0);
    k_h2n<<<GRID1(BB * DD * LL), 0, st>>>(xz, xp, BB, DD, 2 * DD, 0, 0, 0);
    k_dwconv<<<GRID1(BB * DD * LL), 0, st>>>(xp, w.conv_w, w.conv_b, xp2, BB, DD, 1);
    k_build_xst<<<GRID1(BB * LL * DD), 0, st>>>(xp2, xst);
    for (int b = 0; b < BB; ++b)
        for (int k = 0; k < KK; ++k) {
            size_t off = ((size_t)b * KK + k) * LL;
            gemm(st, xst + off * DD, DD, w.x_proj_w + (size_t)k * XPN * DD, DD,
                 nullptr, nullptr, 0, xdb + off * RP, RP, LL, XPN, DD, 0);
        }
    for (int b = 0; b < BB; ++b)
        for (int k = 0; k < KK; ++k) {
            size_t off = ((size_t)b * KK + k) * LL;
            gemm(st, xdb + off * RP, RP, w.dt_proj_w + (size_t)k * DD * RNK, RNK,
                 nullptr, nullptr, 0, dtd + off * DD, DD, LL, DD, RNK, 0);
        }
    k_scan<<<GRID1(BB * KK * DD), 0, st>>>(xst, dtd, w.dt_bias, w.A_logs, xdb, w.Ds, ys);
    k_combine<<<GRID1(BB * LL * DD), 0, st>>>(ys, yc);
    k_ln<<<dim3((ROWS + 7) / 8), dim3(256), 0, st>>>(yc, w.on_w, w.on_b, yl, ROWS, DD);
    k_gate<<<GRID1(ROWS * DD), 0, st>>>(yl, xz, ROWS, DD, 2 * DD, DD);
    gemm(st, yl, DD, w.out_proj_w, DD, nullptr, in, CC, out, CC, ROWS, CC, DD, final_act);
}

static void run_flowmlp(hipStream_t st, const float* x_nchw,
                        const float* convw, const float* convb, const VssW& w,
                        float* out_nchw, float* t0, float* t1, float* t2, float* S)
{
    k_dwconv<<<GRID1(NTOT), 0, st>>>(x_nchw, convw, convb, t0, BB, CC, 0);
    k_n2h<<<GRID1(NTOT), 0, st>>>(t0, t1, BB, CC);
    run_vss(st, t1, t2, w, S, 0);
    k_h2n<<<GRID1(NTOT), 0, st>>>(t2, out_nchw, BB, CC, CC, 0, /*gelu*/1, 0);
}

extern "C" void kernel_launch(void* const* d_in, const int* in_sizes, int n_in,
                              void* d_out, int out_size, void* d_ws, size_t ws_size,
                              hipStream_t stream)
{
    (void)in_sizes; (void)n_in; (void)out_size; (void)ws_size;
    // ---- input pointer map: setup_inputs() dict insertion order, recursive ----
    // 0:x1 1:x2 | fm1: 2 conv_w, 3 conv_b, 4..16 vss | fm2: 17,18, 19..31 vss
    // ca: 32 in_w, 33 in_b, 34 out_w, 35 out_b, 36 ln_w, 37 ln_b
    // 38 norm_w, 39 norm_b, 40 fc1_w, 41 fc1_b, 42 fc2_w, 43 fc2_b
    // mvss1: 44..56 | mvss2: 57..69 | 70 flow_w
    const float* x1 = (const float*)d_in[0];
    const float* x2 = (const float*)d_in[1];
    VssW fm1v  = vss_at(d_in, 4);
    VssW fm2v  = vss_at(d_in, 19);
    VssW mvss1 = vss_at(d_in, 44);
    VssW mvss2 = vss_at(d_in, 57);
    const float* ca_in_w  = (const float*)d_in[32];
    const float* ca_in_b  = (const float*)d_in[33];
    const float* ca_out_w = (const float*)d_in[34];
    const float* ca_out_b = (const float*)d_in[35];
    const float* ca_ln_w  = (const float*)d_in[36];
    const float* ca_ln_b  = (const float*)d_in[37];
    const float* norm_w   = (const float*)d_in[38];
    const float* norm_b   = (const float*)d_in[39];
    const float* fc1_w    = (const float*)d_in[40];
    const float* fc1_b    = (const float*)d_in[41];
    const float* fc2_w    = (const float*)d_in[42];
    const float* fc2_b    = (const float*)d_in[43];
    const float* flow_w   = (const float*)d_in[70];

    float* ws = (float*)d_ws;
    size_t o = 0;
    auto alloc = [&](size_t n) { float* p = ws + o; o += n; return p; };
    float* f1   = alloc(NTOT);
    float* f2   = alloc(NTOT);
    float* x2a  = alloc(NTOT);
    float* tA   = alloc(NTOT);
    float* tB   = alloc(NTOT);
    float* tC   = alloc(NTOT);
    float* qb   = alloc(NTOT);
    float* kb   = alloc(NTOT);
    float* vb   = alloc(NTOT);
    float* flow = alloc(FLOWN);
    float* S    = alloc(SZ_XN + SZ_XZ + 2 * SZ_XP + 3 * SZ_XST + SZ_XDB + 2 * SZ_YC);

    hipStream_t st = stream;

    // ---- x1/x2 feature extraction ----
    run_flowmlp(st, x1, (const float*)d_in[2],  (const float*)d_in[3],  fm1v, f1, tA, tB, tC, S);
    run_flowmlp(st, x2, (const float*)d_in[17], (const float*)d_in[18], fm2v, f2, tA, tB, tC, S);

    // ---- cross-attention align(x2, cond=x1) on raw NCHW-flat (4608,96) rows ----
    k_ln<<<dim3((ROWS + 7) / 8), dim3(256), 0, st>>>(f2, norm_w, norm_b, tA, ROWS, CC); // nx
    k_ln<<<dim3((ROWS + 7) / 8), dim3(256), 0, st>>>(f1, norm_w, norm_b, tB, ROWS, CC); // nc
    gemm(st, tA, CC, ca_in_w,               CC, ca_in_b,          nullptr, 0, qb, CC, ROWS, CC, CC, 0);
    gemm(st, tB, CC, ca_in_w + 1 * CC * CC, CC, ca_in_b + 1 * CC, nullptr, 0, kb, CC, ROWS, CC, CC, 0);
    gemm(st, tB, CC, ca_in_w + 2 * CC * CC, CC, ca_in_b + 2 * CC, nullptr, 0, vb, CC, ROWS, CC, CC, 0);
    k_mha<<<GRID1(LL * NHh), 0, st>>>(qb, kb, vb, tC);
    gemm(st, tC, CC, ca_out_w, CC, ca_out_b, tA, CC, kb, CC, ROWS, CC, CC, 0);  // nx + attn
    k_ln<<<dim3((ROWS + 7) / 8), dim3(256), 0, st>>>(kb, ca_ln_w, ca_ln_b, vb, ROWS, CC); // ca
    k_add<<<GRID1(NTOT), 0, st>>>(f2, vb, x2a, NTOT);

    // ---- MLP branch: LN -> fc1 -> VSS -> gelu -> fc2 -> VSS -> gelu ----
    k_n2h<<<GRID1(NTOT), 0, st>>>(x2a, tA, BB, CC);
    k_ln<<<dim3((ROWS + 7) / 8), dim3(256), 0, st>>>(tA, norm_w, norm_b, tB, ROWS, CC);
    gemm(st, tB, CC, fc1_w, CC, fc1_b, nullptr, 0, tA, CC, ROWS, CC, CC, 0);
    run_vss(st, tA, tB, mvss1, S, /*gelu*/1);
    gemm(st, tB, CC, fc2_w, CC, fc2_b, nullptr, 0, tA, CC, ROWS, CC, CC, 0);
    run_vss(st, tA, tB, mvss2, S, /*gelu*/1);
    k_h2n<<<GRID1(NTOT), 0, st>>>(tB, x2a, BB, CC, CC, 0, 0, /*add*/1);

    // ---- flow head + warp ----
    k_flowconv<<<GRID1(BB * 2 * LL), 0, st>>>(x2a, f1, flow_w, flow);
    float* out = (float*)d_out;
    k_warp<<<GRID1(BB * LL), 0, st>>>(x2a, flow, out);
    k_copy<<<GRID1(FLOWN), 0, st>>>(flow, out + NTOT, FLOWN);
}
